// QLinear_90658169684410
// MI455X (gfx1250) — compile-verified, matrix operand order checked by reading, not run on hardware
//
#include <hip/hip_runtime.h>

typedef unsigned short u16;
typedef unsigned int   u32;
typedef unsigned long long u64;
typedef __attribute__((ext_vector_type(4)))  u32    u32x4;
typedef __attribute__((ext_vector_type(4)))  int    i32x4;
typedef __attribute__((ext_vector_type(8)))  int    i32x8;
typedef __attribute__((ext_vector_type(8)))  __bf16 bf16x8;
typedef __attribute__((ext_vector_type(16))) __bf16 v16bf;
typedef __attribute__((ext_vector_type(8)))  float  v8f;
typedef __attribute__((ext_vector_type(8)))  float  f32x8;

#define K_DIM 4096
#define N_DIM 11008
#define BK    32

union FragU { v16bf v; bf16x8 h[2]; };

// ---- Tensor Data Mover: 2D tile load, descriptor per ISA 8.3/8.4 ----
__device__ __forceinline__ void tdm_load_2d(u32 lds_addr, const void* gaddr,
                                            u32 tensor_d0, u32 tensor_d1,
                                            u64 stride0, u32 tile_d0, u32 tile_d1) {
    u64 ga = (u64)(uintptr_t)gaddr;
    u32x4 g0;
    g0[0] = 1u;                                  // count=1 (valid), user mode
    g0[1] = lds_addr;                            // LDS byte offset
    g0[2] = (u32)ga;                             // global_addr[31:0] -> bits 95:64
    g0[3] = (u32)(ga >> 32) | (2u << 30);        // global_addr[56:32] + type=2
    i32x8 g1;
    g1[0] = (int)(2u << 16);                     // wg_mask=0, data_size=2 (4B)
    g1[1] = (int)((tensor_d0 & 0xFFFFu) << 16);  // tensor_dim0[15:0] @ bits 63:48
    g1[2] = (int)((tensor_d0 >> 16) | ((tensor_d1 & 0xFFFFu) << 16));
    g1[3] = (int)((tensor_d1 >> 16) | (tile_d0 << 16));   // tile_dim0 @ 127:112
    g1[4] = (int)tile_d1;                        // tile_dim1 @ 143:128, tile_dim2=0
    g1[5] = (int)(u32)(stride0 & 0xFFFFFFFFull); // tensor_dim0_stride[31:0]
    g1[6] = (int)(u32)(stride0 >> 32);           // stride[47:32], dim1_stride lo = 0
    g1[7] = 0;
    i32x4 z4 = {0, 0, 0, 0};
    i32x8 z8 = {0, 0, 0, 0, 0, 0, 0, 0};
    __builtin_amdgcn_tensor_load_to_lds(g0, g1, z4, z4, z8, 0);  // 6-arg form
}

__device__ __forceinline__ u32 lds_off(const void* p) {
    return (u32)(uintptr_t)p;   // low 32 bits of generic LDS addr = LDS byte offset
}

__global__ __launch_bounds__(256) void qlinear_wmma_tdm(
    const float* __restrict__ x,
    const int*   __restrict__ pw,
    const float* __restrict__ ws,
    const float* __restrict__ wz,
    float* __restrict__ out,
    int Mtot)
{
    // double-buffered TDM staging
    __shared__ f32x8 Xs[2][128 * 4];   // [buf][m][k] fp32, 16KB each
    __shared__ int   Ws[2][4 * 128];   // [buf][kr][n] packed int4 words, 2KB each
    // bf16 compute tiles, row stride 5*bf16x8 = 40 halfwords (pad 8)
    __shared__ bf16x8 As[128 * 5];     // [m][k]
    __shared__ bf16x8 Bs[128 * 5];     // [n][k]

    const int t    = threadIdx.x;
    const int lane = t & 31;
    const int wave = t >> 5;
    const int wm   = wave & 3;
    const int wn   = wave >> 2;

    const int n0 = blockIdx.x * 128;
    const int m0 = blockIdx.y * 128;

    const int am = t >> 1;             // A convert: row 0..127
    const int ak = (t & 1) * 16;       //            k-half 0/16
    const int bn  = t & 127;           // B dequant: col 0..127
    const int bkr = t >> 7;            //            packed rows {bkr, bkr+2}
    const float scale = ws[n0 + bn];
    const float nzs   = -wz[n0 + bn] * scale;   // (q - zp)*s = fma(q, s, nzs)

    v8f acc[2][4];
    #pragma unroll
    for (int i = 0; i < 2; ++i)
        #pragma unroll
        for (int j = 0; j < 4; ++j)
            acc[i][j] = (v8f)(0.0f);

    const int lhalf = lane >> 4;
    const int lmod  = lane & 15;

    const u32 xs_off0 = lds_off(&Xs[0][0]), xs_off1 = lds_off(&Xs[1][0]);
    const u32 ws_off0 = lds_off(&Ws[0][0]), ws_off1 = lds_off(&Ws[1][0]);

    // prologue: stage k0 = 0 into buffer 0
    if (wave == 0) {
        tdm_load_2d(xs_off0, x + (size_t)m0 * K_DIM, K_DIM, (u32)Mtot,
                    K_DIM, BK, 128);
        tdm_load_2d(ws_off0, pw + (size_t)0 * N_DIM + n0, N_DIM, K_DIM / 8,
                    N_DIM, 128, BK / 8);
    }

    for (int k0 = 0; k0 < K_DIM; k0 += BK) {
        const int cur = (k0 / BK) & 1;

        if (wave == 0) {
            if (k0 + BK < K_DIM) {      // issue next stage, then retire current
                const u32 xo = cur ? xs_off0 : xs_off1;
                const u32 wo = cur ? ws_off0 : ws_off1;
                tdm_load_2d(xo, x + (size_t)m0 * K_DIM + (k0 + BK), K_DIM,
                            (u32)Mtot, K_DIM, BK, 128);
                tdm_load_2d(wo, pw + (size_t)((k0 + BK) >> 3) * N_DIM + n0,
                            N_DIM, K_DIM / 8, N_DIM, 128, BK / 8);
                __builtin_amdgcn_s_wait_tensorcnt(2);
            } else {
                __builtin_amdgcn_s_wait_tensorcnt(0);
            }
        }
        __syncthreads();   // staging buffer `cur` is ready for all waves

        // ---- convert A: fp32 staging -> bf16 tile ----
        {
            f32x8 p0 = Xs[cur][am * 4 + (ak >> 3)];
            f32x8 p1 = Xs[cur][am * 4 + (ak >> 3) + 1];
            As[am * 5 + (ak >> 3)]     = __builtin_convertvector(p0, bf16x8);
            As[am * 5 + (ak >> 3) + 1] = __builtin_convertvector(p1, bf16x8);
        }
        // ---- dequant B: int4 staging -> bf16 tile ----
        #pragma unroll
        for (int rep = 0; rep < 2; ++rep) {
            const int kr = bkr + rep * 2;                 // 0..3
            const u32 w = (u32)Ws[cur][kr * 128 + bn];
            f32x8 f;
            #pragma unroll
            for (int j = 0; j < 8; ++j) {
                const u32 q = (w >> (28 - 4 * j)) & 0xFu; // MSB-first nibbles
                f[j] = fmaf((float)(int)q, scale, nzs);
            }
            Bs[bn * 5 + kr] = __builtin_convertvector(f, bf16x8);
        }
        __syncthreads();   // bf16 tiles published

        // ---- fragments + WMMA ----
        FragU afr[2], bfr[4];
        #pragma unroll
        for (int mt = 0; mt < 2; ++mt) {
            const int row = wm * 32 + mt * 16 + lmod;
            afr[mt].h[0] = As[row * 5 + lhalf];
            afr[mt].h[1] = As[row * 5 + 2 + lhalf];
        }
        #pragma unroll
        for (int nt = 0; nt < 4; ++nt) {
            const int row = wn * 64 + nt * 16 + lmod;
            bfr[nt].h[0] = Bs[row * 5 + lhalf];
            bfr[nt].h[1] = Bs[row * 5 + 2 + lhalf];
        }
        #pragma unroll
        for (int mt = 0; mt < 2; ++mt)
            #pragma unroll
            for (int nt = 0; nt < 4; ++nt)
                acc[mt][nt] = __builtin_amdgcn_wmma_f32_16x16x32_bf16(
                    false, afr[mt].v, false, bfr[nt].v,
                    (short)0, acc[mt][nt], false, false);

        __syncthreads();   // protect As/Bs (and prior staging reads) before reuse
    }

    // ---- write back: C/D layout (VGPR r -> M = r + 8*lhalf, N = lane%16) ----
    #pragma unroll
    for (int mt = 0; mt < 2; ++mt) {
        const int mbase = m0 + wm * 32 + mt * 16 + lhalf * 8;
        #pragma unroll
        for (int nt = 0; nt < 4; ++nt) {
            const int col = n0 + wn * 64 + nt * 16 + lmod;
            float* o = out + (size_t)mbase * N_DIM + col;
            #pragma unroll
            for (int r = 0; r < 8; ++r)
                o[(size_t)r * N_DIM] = acc[mt][nt][r];
        }
    }
}

extern "C" void kernel_launch(void* const* d_in, const int* in_sizes, int n_in,
                              void* d_out, int out_size, void* d_ws, size_t ws_size,
                              hipStream_t stream) {
    const float* x  = (const float*)d_in[0];
    const int*   pw = (const int*)d_in[1];
    const float* ws = (const float*)d_in[2];
    const float* wz = (const float*)d_in[3];
    float* out = (float*)d_out;

    const int M = in_sizes[0] / K_DIM;      // 8192
    dim3 grid(N_DIM / 128, M / 128);        // 86 x 64
    dim3 block(256);
    qlinear_wmma_tdm<<<grid, block, 0, stream>>>(x, pw, ws, wz, out, M);
}
